// Locoformer_31602369364496
// MI455X (gfx1250) — matrix operand order repaired and checked by reading, not compiled, not run on hardware
//
#include <hip/hip_runtime.h>
#include <hip/hip_bf16.h>
#include <math.h>

// ---------------- problem constants ----------------
#define S_LEN   2048
#define D_MODEL 1024
#define N_HEADS 16
#define D_HEAD  64
#define HD_     1024        // N_HEADS * D_HEAD
#define F_IN    2730
#define F_PAD   2752        // pad F to multiple of 64
#define N1      5460        // 2*F_IN
#define N1_PAD  5504        // pad to multiple of 128 (43*128)
#define WIN     1024
#define S_PAD   (S_LEN + 64)
#define NEG_BIG (-3.0e38f)

typedef __bf16 bf16_t;
typedef __bf16 v16bf __attribute__((ext_vector_type(16)));
typedef float  v8f   __attribute__((ext_vector_type(8)));
typedef int    v4i_  __attribute__((ext_vector_type(4)));

union FragB { v16bf v; uint4 u[2]; };

__device__ __forceinline__ bf16_t f2bf(float f) {
  union { float f; unsigned u; } a; a.f = f;
  unsigned r = a.u + 0x7FFFu + ((a.u >> 16) & 1u);
  union { unsigned short s; bf16_t b; } o; o.s = (unsigned short)(r >> 16);
  return o.b;
}
__device__ __forceinline__ float sigmoidf_(float x) { return 1.0f / (1.0f + __expf(-x)); }

// full-wave xor-butterfly step via ds_swizzle (group-of-32: and=0x1f, or=0, xor=XORM)
template<int XORM>
__device__ __forceinline__ float swz_xor(float x) {
  return __int_as_float(
      __builtin_amdgcn_ds_swizzle(__float_as_int(x), (XORM << 10) | 0x1f));
}

// ---------------- gfx1250 async global->LDS copy (ASYNCcnt path) ----------------
#if defined(__has_builtin)
#if __has_builtin(__builtin_amdgcn_global_load_async_to_lds_b128)
#define HAVE_ASYNC_LDS 1
#endif
#endif
#ifndef HAVE_ASYNC_LDS
#define HAVE_ASYNC_LDS 0
#endif

typedef __attribute__((address_space(1))) v4i_* gas_v4i;
typedef __attribute__((address_space(3))) v4i_* las_v4i;

__device__ __forceinline__ void async_cp16(void* lds, const void* g) {
#if HAVE_ASYNC_LDS
  __builtin_amdgcn_global_load_async_to_lds_b128(
      (gas_v4i)(void*)g, (las_v4i)lds, 0, 0);
#else
  *(uint4*)lds = *(const uint4*)g;
#endif
}
template<int N>
__device__ __forceinline__ void async_wait() {
#if HAVE_ASYNC_LDS
#if __has_builtin(__builtin_amdgcn_s_wait_asynccnt)
  __builtin_amdgcn_s_wait_asynccnt(N);
#else
  asm volatile("s_wait_asynccnt %0" :: "n"(N) : "memory");
#endif
#endif
}

// ---------------- cooperative WMMA GEMM with async-LDS double buffering ----------
// Block = 256 threads (8 waves, 4x2). Block tile 128x128, wave tile 32x64.
// A[M x K] bf16 row-major; Bt[Npad x K] bf16 (transposed weights); C[M x Npad] f32.
#define LDS_STR 40   // padded row stride (elements) to avoid ds bank conflicts
__global__ __launch_bounds__(256) void gemm_lds_k(
    const bf16_t* __restrict__ A, const bf16_t* __restrict__ Bt,
    float* __restrict__ C, const float* __restrict__ bias, int Nbias,
    int Npad, int K)
{
  __shared__ __attribute__((aligned(16))) bf16_t ldsA[2][128 * LDS_STR];
  __shared__ __attribute__((aligned(16))) bf16_t ldsB[2][128 * LDS_STR];

  const int tid  = threadIdx.x;
  const int lane = tid & 31, wave = tid >> 5;
  const int hf   = lane >> 4, l16 = lane & 15;
  const int wm   = wave >> 1;          // 0..3 : M position
  const int wn   = wave & 1;           // 0..1 : N position
  const int m0   = blockIdx.y * 128;
  const int n0   = blockIdx.x * 128;

  v8f acc[2][4];
#pragma unroll
  for (int mt = 0; mt < 2; ++mt)
#pragma unroll
    for (int nt = 0; nt < 4; ++nt)
#pragma unroll
      for (int r = 0; r < 8; ++r) acc[mt][nt][r] = 0.0f;

  // cooperative copy assignment: each thread moves one 16B chunk of A and of B
  const int rowC = tid >> 1;            // 0..127
  const int ch   = (tid & 1) * 16;      // 0 or 16 (elements)
  const bf16_t* gA = A  + (size_t)(m0 + rowC) * K + ch;
  const bf16_t* gB = Bt + (size_t)(n0 + rowC) * K + ch;

  auto issue = [&](int buf, int kk) {
    async_cp16(&ldsA[buf][rowC * LDS_STR + ch], gA + kk);
    async_cp16(&ldsB[buf][rowC * LDS_STR + ch], gB + kk);
  };

  issue(0, 0);
  int buf = 0;
  for (int kk = 0; kk < K; kk += 32) {
    const bool more = (kk + 32) < K;
    if (more) issue(buf ^ 1, kk + 32);
    if (more) async_wait<2>(); else async_wait<0>();
    __syncthreads();

    const bf16_t* tA = &ldsA[buf][(wm * 32) * LDS_STR];
    const bf16_t* tB = &ldsB[buf][(wn * 64) * LDS_STR];
    FragB a[2];
#pragma unroll
    for (int mt = 0; mt < 2; ++mt) {
      const bf16_t* ap = tA + (mt * 16 + l16) * LDS_STR;
      a[mt].u[0] = *(const uint4*)(ap + hf * 8);
      a[mt].u[1] = *(const uint4*)(ap + 16 + hf * 8);
    }
#pragma unroll
    for (int nt = 0; nt < 4; ++nt) {
      FragB b;
      const bf16_t* bp = tB + (nt * 16 + l16) * LDS_STR + hf * 8;
      b.u[0] = *(const uint4*)(bp);
      b.u[1] = *(const uint4*)(bp + 16);
#pragma unroll
      for (int mt = 0; mt < 2; ++mt)
        acc[mt][nt] = __builtin_amdgcn_wmma_f32_16x16x32_bf16(
            false, a[mt].v, false, b.v, (short)0, acc[mt][nt], false, false);
    }
    __syncthreads();
    buf ^= 1;
  }

#pragma unroll
  for (int mt = 0; mt < 2; ++mt)
#pragma unroll
    for (int nt = 0; nt < 4; ++nt) {
      int col = n0 + wn * 64 + nt * 16 + l16;
      float bv = 0.0f;
      if (bias != nullptr && col < Nbias) bv = bias[col];
#pragma unroll
      for (int r = 0; r < 8; ++r) {
        int row = m0 + wm * 32 + mt * 16 + r + hf * 8;
        C[(size_t)row * Npad + col] = acc[mt][nt][r] + bv;
      }
    }
}

// ---------------- small-N per-wave WMMA GEMM (for H=16 outputs) ----------------
template<int MT, int NT>
__global__ __launch_bounds__(32) void gemm_wmma_k(
    const bf16_t* __restrict__ A, const bf16_t* __restrict__ Bt,
    float* __restrict__ C, const float* __restrict__ bias, int Nbias,
    int Npad, int K)
{
  const int lane = threadIdx.x;
  const int hf   = lane >> 4;
  const int l16  = lane & 15;
  const int m0   = blockIdx.y * (16 * MT);
  const int n0   = blockIdx.x * (16 * NT);

  const bf16_t* arow[MT];
#pragma unroll
  for (int mt = 0; mt < MT; ++mt)
    arow[mt] = A + (size_t)(m0 + mt * 16 + l16) * K;

  v8f acc[MT][NT];
#pragma unroll
  for (int mt = 0; mt < MT; ++mt)
#pragma unroll
    for (int nt = 0; nt < NT; ++nt)
#pragma unroll
      for (int r = 0; r < 8; ++r) acc[mt][nt][r] = 0.0f;

  for (int kk = 0; kk < K; kk += 32) {
    FragB a[MT];
#pragma unroll
    for (int mt = 0; mt < MT; ++mt) {
      a[mt].u[0] = *(const uint4*)(arow[mt] + kk + hf * 8);
      a[mt].u[1] = *(const uint4*)(arow[mt] + kk + 16 + hf * 8);
      if (kk + 128 < K) __builtin_prefetch(arow[mt] + kk + 128, 0, 1);
    }
#pragma unroll
    for (int nt = 0; nt < NT; ++nt) {
      FragB b;
      const bf16_t* bp = Bt + (size_t)(n0 + nt * 16 + l16) * K + kk + hf * 8;
      b.u[0] = *(const uint4*)(bp);
      b.u[1] = *(const uint4*)(bp + 16);
#pragma unroll
      for (int mt = 0; mt < MT; ++mt)
        acc[mt][nt] = __builtin_amdgcn_wmma_f32_16x16x32_bf16(
            false, a[mt].v, false, b.v, (short)0, acc[mt][nt], false, false);
    }
  }

#pragma unroll
  for (int mt = 0; mt < MT; ++mt)
#pragma unroll
    for (int nt = 0; nt < NT; ++nt) {
      int col = n0 + nt * 16 + l16;
      float bv = 0.0f;
      if (bias != nullptr && col < Nbias) bv = bias[col];
#pragma unroll
      for (int r = 0; r < 8; ++r) {
        int row = m0 + mt * 16 + r + hf * 8;
        C[(size_t)row * Npad + col] = acc[mt][nt][r] + bv;
      }
    }
}

// ---------------- flash attention, windowed, one wave per (head, 16-query block) ----
__global__ __launch_bounds__(32) void attn_wmma_k(
    const bf16_t* __restrict__ Qb,   // [S x HD], rope+scale applied
    const bf16_t* __restrict__ Kb,   // [S x HD], rope applied
    const bf16_t* __restrict__ Vt,   // [HD x S_PAD], transposed, value-mixed
    float* __restrict__ O)           // [S x HD]
{
  __shared__ __attribute__((aligned(16))) bf16_t pt[16 * 32];
  const int lane = threadIdx.x;
  const int hf   = lane >> 4;
  const int l16  = lane & 15;
  const int i0   = blockIdx.x * 16;
  const int hh   = blockIdx.y;

  FragB q0, q1;
  {
    const bf16_t* qrow = Qb + (size_t)(i0 + l16) * HD_ + hh * D_HEAD;
    q0.u[0] = *(const uint4*)(qrow + hf * 16);
    q0.u[1] = *(const uint4*)(qrow + hf * 16 + 8);
    q1.u[0] = *(const uint4*)(qrow + 32 + hf * 16);
    q1.u[1] = *(const uint4*)(qrow + 40 + hf * 16);
  }

  FragB ones;
  {
    const unsigned oc = 0x3F803F80u;   // bf16 1.0 pair
    ones.u[0] = make_uint4(oc, oc, oc, oc);
    ones.u[1] = make_uint4(oc, oc, oc, oc);
  }

  v8f o[4], osum;
#pragma unroll
  for (int t = 0; t < 4; ++t)
#pragma unroll
    for (int r = 0; r < 8; ++r) o[t][r] = 0.0f;
#pragma unroll
  for (int r = 0; r < 8; ++r) osum[r] = 0.0f;
  float mcur = NEG_BIG;

  int jstart = i0 - WIN; if (jstart < 0) jstart = 0;
  for (int j0 = jstart & ~31; j0 <= i0 + 15; j0 += 32) {
    v8f s0 = {0,0,0,0,0,0,0,0}, s1 = {0,0,0,0,0,0,0,0};
    {
      int jA = j0 + l16;       if (jA > S_LEN - 1) jA = S_LEN - 1;
      const bf16_t* kr = Kb + (size_t)jA * HD_ + hh * D_HEAD;
      FragB b;
      b.u[0] = *(const uint4*)(kr + hf * 16);
      b.u[1] = *(const uint4*)(kr + hf * 16 + 8);
      s0 = __builtin_amdgcn_wmma_f32_16x16x32_bf16(false, q0.v, false, b.v, (short)0, s0, false, false);
      b.u[0] = *(const uint4*)(kr + 32 + hf * 16);
      b.u[1] = *(const uint4*)(kr + 40 + hf * 16);
      s0 = __builtin_amdgcn_wmma_f32_16x16x32_bf16(false, q1.v, false, b.v, (short)0, s0, false, false);

      int jB = j0 + 16 + l16;  if (jB > S_LEN - 1) jB = S_LEN - 1;
      kr = Kb + (size_t)jB * HD_ + hh * D_HEAD;
      b.u[0] = *(const uint4*)(kr + hf * 16);
      b.u[1] = *(const uint4*)(kr + hf * 16 + 8);
      s1 = __builtin_amdgcn_wmma_f32_16x16x32_bf16(false, q0.v, false, b.v, (short)0, s1, false, false);
      b.u[0] = *(const uint4*)(kr + 32 + hf * 16);
      b.u[1] = *(const uint4*)(kr + 40 + hf * 16);
      s1 = __builtin_amdgcn_wmma_f32_16x16x32_bf16(false, q1.v, false, b.v, (short)0, s1, false, false);
    }

    bool mA[8], mB[8];
    float tmax = NEG_BIG;
#pragma unroll
    for (int r = 0; r < 8; ++r) {
      int qi = i0 + r + 8 * hf;
      int jA = j0 + l16, jB = jA + 16;
      mA[r] = (jA > qi) || (jA < qi - WIN);
      mB[r] = (jB > qi) || (jB < qi - WIN);
      float va = mA[r] ? NEG_BIG : s0[r];
      float vb = mB[r] ? NEG_BIG : s1[r];
      tmax = fmaxf(tmax, fmaxf(va, vb));
    }
    tmax = fmaxf(tmax, swz_xor<1>(tmax));
    tmax = fmaxf(tmax, swz_xor<2>(tmax));
    tmax = fmaxf(tmax, swz_xor<4>(tmax));
    tmax = fmaxf(tmax, swz_xor<8>(tmax));
    tmax = fmaxf(tmax, swz_xor<16>(tmax));

    float nm    = fmaxf(mcur, tmax);
    float alpha = __expf(mcur - nm);
    mcur = nm;

#pragma unroll
    for (int r = 0; r < 8; ++r) {
      float pa = mA[r] ? 0.0f : __expf(s0[r] - nm);
      float pb = mB[r] ? 0.0f : __expf(s1[r] - nm);
      pt[(r + 8 * hf) * 32 + l16]      = f2bf(pa);
      pt[(r + 8 * hf) * 32 + 16 + l16] = f2bf(pb);
    }
#pragma unroll
    for (int t = 0; t < 4; ++t)
#pragma unroll
      for (int r = 0; r < 8; ++r) o[t][r] *= alpha;
#pragma unroll
    for (int r = 0; r < 8; ++r) osum[r] *= alpha;
    __syncthreads();

    FragB p;
    p.u[0] = *(const uint4*)(&pt[l16 * 32 + hf * 8]);
    p.u[1] = *(const uint4*)(&pt[l16 * 32 + 16 + hf * 8]);
    osum = __builtin_amdgcn_wmma_f32_16x16x32_bf16(false, p.v, false, ones.v, (short)0, osum, false, false);
#pragma unroll
    for (int t = 0; t < 4; ++t) {
      FragB b;
      const bf16_t* vp = Vt + (size_t)(hh * D_HEAD + t * 16 + l16) * S_PAD + j0 + hf * 8;
      b.u[0] = *(const uint4*)(vp);
      b.u[1] = *(const uint4*)(vp + 16);
      o[t] = __builtin_amdgcn_wmma_f32_16x16x32_bf16(false, p.v, false, b.v, (short)0, o[t], false, false);
    }
    __syncthreads();
  }

#pragma unroll
  for (int t = 0; t < 4; ++t)
#pragma unroll
    for (int r = 0; r < 8; ++r) {
      int row = i0 + r + 8 * hf;
      O[(size_t)row * HD_ + hh * D_HEAD + t * 16 + l16] = o[t][r] / osum[r];
    }
}

// ---------------- RMSNorm (one block per row) ----------------
__global__ __launch_bounds__(256) void rmsnorm_k(
    const float* __restrict__ x, const float* __restrict__ w,
    bf16_t* __restrict__ obf, float* __restrict__ of32, int D)
{
  __shared__ float red[8];
  __shared__ float rscale;
  int row = blockIdx.x;
  const float* xr = x + (size_t)row * D;
  float s = 0.0f;
  for (int c = threadIdx.x; c < D; c += 256) { float v = xr[c]; s += v * v; }
  s += swz_xor<16>(s); s += swz_xor<8>(s); s += swz_xor<4>(s);
  s += swz_xor<2>(s);  s += swz_xor<1>(s);
  if ((threadIdx.x & 31) == 0) red[threadIdx.x >> 5] = s;
  __syncthreads();
  if (threadIdx.x == 0) {
    float t = 0.0f;
    for (int i = 0; i < 8; ++i) t += red[i];
    rscale = rsqrtf(t / (float)D + 1.1920929e-7f);
  }
  __syncthreads();
  float r = rscale;
  for (int c = threadIdx.x; c < D; c += 256) {
    float v = xr[c] * r * w[c];
    if (obf)  obf[(size_t)row * D + c]  = f2bf(v);
    if (of32) of32[(size_t)row * D + c] = v;
  }
}

// ---------------- weight transpose + f32->bf16 (zero padded) ----------------
__global__ __launch_bounds__(256) void convT_k(
    const float* __restrict__ src, bf16_t* __restrict__ dst,
    int K, int N, int Kpad, int Npad)
{
  size_t idx = (size_t)blockIdx.x * 256 + threadIdx.x;
  size_t tot = (size_t)Npad * Kpad;
  if (idx >= tot) return;
  int k = (int)(idx % Kpad);
  int n = (int)(idx / Kpad);
  float v = (n < N && k < K) ? src[(size_t)k * N + n] : 0.0f;
  dst[idx] = f2bf(v);
}

// ---------------- RoPE (+scale) f32 -> bf16 ----------------
__global__ __launch_bounds__(256) void rope_k(
    const float* __restrict__ in, int istride, bf16_t* __restrict__ out, float scale)
{
  int idx = blockIdx.x * 256 + threadIdx.x;           // over S*HD/2
  if (idx >= S_LEN * HD_ / 2) return;
  int s = idx / (HD_ / 2);
  int p = idx % (HD_ / 2);
  int head = p >> 5;
  int i = p & 31;
  float inv = __expf((float)i * -0.28782313662f);     // 10000^(-2i/64)
  float ang = (float)s * inv;
  float cs, sn;
  __sincosf(ang, &sn, &cs);
  const float* base = in + (size_t)s * istride + head * 64;
  float x1 = base[2 * i], x2 = base[2 * i + 1];
  bf16_t* ob = out + (size_t)s * HD_ + head * 64;
  ob[2 * i]     = f2bf((x1 * cs - x2 * sn) * scale);
  ob[2 * i + 1] = f2bf((x2 * cs + x1 * sn) * scale);
}

// ---------------- value-residual mix + transpose to [HD x S_PAD] bf16 ----------------
__global__ __launch_bounds__(256) void mixv_k(
    const float* __restrict__ kv, const float* __restrict__ mixl,
    float* __restrict__ vres, bf16_t* __restrict__ vT, int layer)
{
  int idx = blockIdx.x * 256 + threadIdx.x;           // over S*HD
  if (idx >= S_LEN * HD_) return;
  int s = idx / HD_, d = idx % HD_;
  float v = kv[(size_t)s * (2 * HD_) + HD_ + d];
  if (layer == 0) {
    vres[idx] = v;
  } else {
    float m = sigmoidf_(mixl[s * N_HEADS + (d >> 6)]);
    float vr = vres[idx];
    v = v + m * (vr - v);
  }
  vT[(size_t)d * S_PAD + s] = f2bf(v);
}

// ---------------- head gating ----------------
__global__ __launch_bounds__(256) void gate_k(
    const float* __restrict__ attno, const float* __restrict__ gatel,
    bf16_t* __restrict__ gb)
{
  int idx = blockIdx.x * 256 + threadIdx.x;
  if (idx >= S_LEN * HD_) return;
  int s = idx / HD_, d = idx % HD_;
  float g = sigmoidf_(gatel[s * N_HEADS + (d >> 6)]);
  gb[idx] = f2bf(attno[idx] * g);
}

// ---------------- MLP activation ----------------
__global__ __launch_bounds__(256) void act_k(
    const float* __restrict__ h, bf16_t* __restrict__ act)
{
  int idx = blockIdx.x * 256 + threadIdx.x;           // over S*F_PAD
  if (idx >= S_LEN * F_PAD) return;
  int s = idx / F_PAD, c = idx % F_PAD;
  float r = 0.0f;
  if (c < F_IN) {
    float a = h[(size_t)s * N1_PAD + c];
    float g = h[(size_t)s * N1_PAD + F_IN + c];
    float ge = 0.5f * g * (1.0f + erff(g * 0.70710678118f));
    r = a * ge;
  }
  act[idx] = f2bf(r);
}

// ---------------- residual add ----------------
__global__ __launch_bounds__(256) void add_k(
    float* __restrict__ x, const float* __restrict__ y, int n)
{
  int idx = blockIdx.x * 256 + threadIdx.x;
  if (idx < n) x[idx] += y[idx];
}

// ---------------- orchestration ----------------
extern "C" void kernel_launch(void* const* d_in, const int* in_sizes, int n_in,
                              void* d_out, int out_size, void* d_ws, size_t ws_size,
                              hipStream_t stream) {
  (void)in_sizes; (void)n_in; (void)out_size; (void)ws_size;
  const float* x_in   = (const float*)d_in[0];
  const float* norm1w = (const float*)d_in[1];
  const float* wq     = (const float*)d_in[2];
  const float* wkv    = (const float*)d_in[3];
  const float* wo     = (const float*)d_in[4];
  const float* wg     = (const float*)d_in[5];
  const float* wmix   = (const float*)d_in[6];
  const float* norm2w = (const float*)d_in[7];
  const float* w1     = (const float*)d_in[8];
  const float* b1     = (const float*)d_in[9];
  const float* w2     = (const float*)d_in[10];
  const float* b2     = (const float*)d_in[11];
  const float* fnw    = (const float*)d_in[12];

  char* p = (char*)d_ws;
  auto alloc = [&](size_t bytes) -> char* {
    char* r = p; p += (bytes + 255) & ~(size_t)255; return r;
  };
  float*  xb    = (float*) alloc(sizeof(float)  * S_LEN * D_MODEL);
  bf16_t* tbf   = (bf16_t*)alloc(sizeof(bf16_t) * S_LEN * D_MODEL);
  float*  qf    = (float*) alloc(sizeof(float)  * S_LEN * HD_);
  float*  kvf   = (float*) alloc(sizeof(float)  * S_LEN * 2 * HD_);
  bf16_t* qb    = (bf16_t*)alloc(sizeof(bf16_t) * S_LEN * HD_);
  bf16_t* kb    = (bf16_t*)alloc(sizeof(bf16_t) * S_LEN * HD_);
  float*  mixl  = (float*) alloc(sizeof(float)  * S_LEN * N_HEADS);
  float*  gatel = (float*) alloc(sizeof(float)  * S_LEN * N_HEADS);
  float*  vres  = (float*) alloc(sizeof(float)  * S_LEN * HD_);
  bf16_t* vT    = (bf16_t*)alloc(sizeof(bf16_t) * HD_ * S_PAD);
  float*  attno = (float*) alloc(sizeof(float)  * S_LEN * HD_);
  bf16_t* gbuf  = (bf16_t*)alloc(sizeof(bf16_t) * S_LEN * HD_);
  float*  hbuf  = (float*) alloc(sizeof(float)  * S_LEN * N1_PAD);
  bf16_t* actb  = (bf16_t*)alloc(sizeof(bf16_t) * S_LEN * F_PAD);
  float*  tmp   = (float*) alloc(sizeof(float)  * S_LEN * D_MODEL);
  bf16_t* wqT   = (bf16_t*)alloc(sizeof(bf16_t) * HD_ * D_MODEL);
  bf16_t* wkvT  = (bf16_t*)alloc(sizeof(bf16_t) * 2 * HD_ * D_MODEL);
  bf16_t* wgT   = (bf16_t*)alloc(sizeof(bf16_t) * 16 * D_MODEL);
  bf16_t* wmixT = (bf16_t*)alloc(sizeof(bf16_t) * 16 * D_MODEL);
  bf16_t* woT   = (bf16_t*)alloc(sizeof(bf16_t) * D_MODEL * HD_);
  bf16_t* w1T   = (bf16_t*)alloc(sizeof(bf16_t) * N1_PAD * D_MODEL);
  bf16_t* w2T   = (bf16_t*)alloc(sizeof(bf16_t) * D_MODEL * F_PAD);

  (void)hipMemcpyAsync(xb, x_in, sizeof(float) * S_LEN * D_MODEL,
                       hipMemcpyDeviceToDevice, stream);

  const int EW = 256;
  auto blocks = [](size_t n) { return (unsigned)((n + 255) / 256); };

  for (int l = 0; l < 2; ++l) {
    const float* wq_l   = wq   + (size_t)l * D_MODEL * HD_;
    const float* wkv_l  = wkv  + (size_t)l * D_MODEL * 2 * HD_;
    const float* wo_l   = wo   + (size_t)l * HD_ * D_MODEL;
    const float* wg_l   = wg   + (size_t)l * D_MODEL * N_HEADS;
    const float* wmix_l = wmix + (size_t)l * D_MODEL * N_HEADS;
    const float* w1_l   = w1   + (size_t)l * D_MODEL * N1;
    const float* b1_l   = b1   + (size_t)l * N1;
    const float* w2_l   = w2   + (size_t)l * F_IN * D_MODEL;
    const float* b2_l   = b2   + (size_t)l * D_MODEL;

    rmsnorm_k<<<S_LEN, EW, 0, stream>>>(xb, norm1w + l * D_MODEL, tbf, nullptr, D_MODEL);

    convT_k<<<blocks((size_t)HD_ * D_MODEL), EW, 0, stream>>>(wq_l,   wqT,   D_MODEL, HD_,     D_MODEL, HD_);
    convT_k<<<blocks((size_t)2 * HD_ * D_MODEL), EW, 0, stream>>>(wkv_l, wkvT, D_MODEL, 2 * HD_, D_MODEL, 2 * HD_);
    convT_k<<<blocks((size_t)16 * D_MODEL), EW, 0, stream>>>(wg_l,   wgT,   D_MODEL, N_HEADS, D_MODEL, 16);
    convT_k<<<blocks((size_t)16 * D_MODEL), EW, 0, stream>>>(wmix_l, wmixT, D_MODEL, N_HEADS, D_MODEL, 16);
    convT_k<<<blocks((size_t)D_MODEL * HD_), EW, 0, stream>>>(wo_l,  woT,   HD_,     D_MODEL, HD_,     D_MODEL);
    convT_k<<<blocks((size_t)N1_PAD * D_MODEL), EW, 0, stream>>>(w1_l, w1T,  D_MODEL, N1,      D_MODEL, N1_PAD);
    convT_k<<<blocks((size_t)D_MODEL * F_PAD), EW, 0, stream>>>(w2_l, w2T,  F_IN,    D_MODEL, F_PAD,   D_MODEL);

    // projections: cooperative async-LDS GEMM (128x128 block tile)
    gemm_lds_k<<<dim3(HD_ / 128, S_LEN / 128), 256, 0, stream>>>(tbf, wqT,  qf,  nullptr, 0, HD_,     D_MODEL);
    gemm_lds_k<<<dim3(2 * HD_ / 128, S_LEN / 128), 256, 0, stream>>>(tbf, wkvT, kvf, nullptr, 0, 2 * HD_, D_MODEL);
    gemm_wmma_k<2,1><<<dim3(1, S_LEN / 32), 32, 0, stream>>>(tbf, wmixT, mixl, nullptr, 0, 16, D_MODEL);
    gemm_wmma_k<2,1><<<dim3(1, S_LEN / 32), 32, 0, stream>>>(tbf, wgT,   gatel, nullptr, 0, 16, D_MODEL);

    rope_k<<<blocks((size_t)S_LEN * HD_ / 2), EW, 0, stream>>>(qf, HD_, qb, 0.125f);
    rope_k<<<blocks((size_t)S_LEN * HD_ / 2), EW, 0, stream>>>(kvf, 2 * HD_, kb, 1.0f);

    (void)hipMemsetAsync(vT, 0, sizeof(bf16_t) * HD_ * S_PAD, stream);
    mixv_k<<<blocks((size_t)S_LEN * HD_), EW, 0, stream>>>(kvf, mixl, vres, vT, l);

    attn_wmma_k<<<dim3(S_LEN / 16, N_HEADS), 32, 0, stream>>>(qb, kb, vT, attno);

    gate_k<<<blocks((size_t)S_LEN * HD_), EW, 0, stream>>>(attno, gatel, gbuf);
    gemm_lds_k<<<dim3(D_MODEL / 128, S_LEN / 128), 256, 0, stream>>>(gbuf, woT, tmp, nullptr, 0, D_MODEL, HD_);
    add_k<<<blocks((size_t)S_LEN * D_MODEL), EW, 0, stream>>>(xb, tmp, S_LEN * D_MODEL);

    rmsnorm_k<<<S_LEN, EW, 0, stream>>>(xb, norm2w + l * D_MODEL, tbf, nullptr, D_MODEL);
    gemm_lds_k<<<dim3(N1_PAD / 128, S_LEN / 128), 256, 0, stream>>>(tbf, w1T, hbuf, b1_l, N1, N1_PAD, D_MODEL);
    act_k<<<blocks((size_t)S_LEN * F_PAD), EW, 0, stream>>>(hbuf, actb);
    gemm_lds_k<<<dim3(D_MODEL / 128, S_LEN / 128), 256, 0, stream>>>(actb, w2T, tmp, b2_l, D_MODEL, D_MODEL, F_PAD);
    add_k<<<blocks((size_t)S_LEN * D_MODEL), EW, 0, stream>>>(xb, tmp, S_LEN * D_MODEL);
  }

  rmsnorm_k<<<S_LEN, EW, 0, stream>>>(xb, fnw, nullptr, (float*)d_out, D_MODEL);
}
